// TensorFlowModel_61375082660134
// MI455X (gfx1250) — matrix operand order repaired and checked by reading, not compiled, not run on hardware
//
#include <hip/hip_runtime.h>
#include <math.h>

typedef __attribute__((ext_vector_type(16))) _Float16 v16h;
typedef __attribute__((ext_vector_type(8)))  _Float16 v8h;
typedef __attribute__((ext_vector_type(8)))  float    v8f;

#define BQ    4096      // query batch
#define BUFN  4096      // sliding buffer length
#define CATN  8192      // BUFN + BQ
#define CATP  8448      // cat rows padded so wide col tiles never read OOB
#define DIMD  784       // 28*28
#define KP    800       // 784 padded to multiple of 32
#define TOPK  50
#define EPSD  1e-12f

// ---------------------------------------------------------------------------
// Kernel 1: f16 hi/lo split of cat = [buf; x] (zero-padded to KP cols and to
// CATP rows) + per-row squared L2 norms (fp32 from original data).
// ---------------------------------------------------------------------------
__global__ void k_convert(const float* __restrict__ x, const float* __restrict__ buf,
                          _Float16* __restrict__ chi, _Float16* __restrict__ clo,
                          float* __restrict__ cnorm) {
  const int r = blockIdx.x;
  float ss = 0.f;
  for (int c = threadIdx.x; c < KP; c += blockDim.x) {
    float v = 0.f;
    if (r < CATN && c < DIMD)
      v = (r < BUFN) ? buf[(size_t)r * DIMD + c] : x[(size_t)(r - BUFN) * DIMD + c];
    _Float16 h = (_Float16)v;
    chi[(size_t)r * KP + c] = h;
    clo[(size_t)r * KP + c] = (_Float16)(v - (float)h);
    ss += v * v;
  }
  __shared__ float red[256];
  red[threadIdx.x] = ss;
  __syncthreads();
  for (int s = 128; s > 0; s >>= 1) {
    if (threadIdx.x < (unsigned)s) red[threadIdx.x] += red[threadIdx.x + s];
    __syncthreads();
  }
  if (threadIdx.x == 0) cnorm[r] = red[0];
}

// ---------------------------------------------------------------------------
// Kernel 2: banded distance GEMM, f16 hi/lo split (3 WMMAs per product).
// Wave computes a 32x32 output tile (2x2 grid of 16x16 WMMA tiles, 4 f32
// accumulators) -> 24 FLOP/byte of fragment traffic without register spills.
// Block = 8 waves in a 2x4 arrangement = 64x128 macro tile. Band:
// c0 = r0 + 128*blockIdx.x, blockIdx.x in [0,33). dist scattered into
// windows[i][j], j = c - i.
// ---------------------------------------------------------------------------
__global__ void __launch_bounds__(256, 1)
k_band_gemm(const _Float16* __restrict__ chi,
            const _Float16* __restrict__ clo,
            const float* __restrict__ cnorm,
            float* __restrict__ windows) {
  const int lane = threadIdx.x & 31;
  const int wave = threadIdx.x >> 5;
  const int mi2  = wave >> 2;                     // 0..1: 32-row group
  const int ni2  = wave & 3;                      // 0..3: 32-col group
  const int r0   = blockIdx.y << 6;               // block row base (64 rows)
  const int cb   = r0 + (blockIdx.x << 7);        // block col base (128 cols)
  const int rw   = r0 + (mi2 << 5);               // wave row base (32 rows)
  const int cw   = cb + (ni2 << 5);               // wave col base (32 cols)

  const int hihalf = lane >> 4;                   // 0 or 1
  const int l15    = lane & 15;
  const int akoff  = hihalf << 3;                 // A k-offset 0 / 8
  const int bkoff  = hihalf << 4;                 // B k-offset 0 / 16

  // A fragment rows (xf rows = cat rows BUFN + i)
  size_t arow[2];
  arow[0] = (size_t)(BUFN + rw + l15) * KP;
  arow[1] = (size_t)(BUFN + rw + 16 + l15) * KP;
  // B fragment rows (cat rows; padded region is zeros)
  size_t brow[2];
  brow[0] = (size_t)(cw + l15) * KP;
  brow[1] = (size_t)(cw + 16 + l15) * KP;

  v8f acc[2][2] = {};

  for (int kb = 0; kb < KP; kb += 32) {
    v16h ah[2], al[2], bh[2], bl[2];
#pragma unroll
    for (int mi = 0; mi < 2; ++mi) {
      // A 16x32 layout: lane<16 holds k {0..7,16..23}; lane>=16 {8..15,24..31}
      v8h h0 = *(const v8h*)(chi + arow[mi] + kb + akoff);
      v8h h1 = *(const v8h*)(chi + arow[mi] + kb + akoff + 16);
      v8h l0 = *(const v8h*)(clo + arow[mi] + kb + akoff);
      v8h l1 = *(const v8h*)(clo + arow[mi] + kb + akoff + 16);
      ah[mi] = __builtin_shufflevector(h0, h1, 0,1,2,3,4,5,6,7,8,9,10,11,12,13,14,15);
      al[mi] = __builtin_shufflevector(l0, l1, 0,1,2,3,4,5,6,7,8,9,10,11,12,13,14,15);
    }
#pragma unroll
    for (int ni = 0; ni < 2; ++ni) {
      // B 32x16 layout: lane n<16 -> col n, k {0..7,16..23}; lane n+16 -> k {8..15,24..31}
      v8h bh0 = *(const v8h*)(chi + brow[ni] + kb + akoff);
      v8h bh1 = *(const v8h*)(chi + brow[ni] + kb + akoff + 16);
      v8h bl0 = *(const v8h*)(clo + brow[ni] + kb + akoff);
      v8h bl1 = *(const v8h*)(clo + brow[ni] + kb + akoff + 16);
      bh[ni] = __builtin_shufflevector(bh0, bh1, 0,1,2,3,4,5,6,7,8,9,10,11,12,13,14,15);
      bl[ni] = __builtin_shufflevector(bl0, bl1, 0,1,2,3,4,5,6,7,8,9,10,11,12,13,14,15);
    }
#pragma unroll
    for (int mi = 0; mi < 2; ++mi)
#pragma unroll
      for (int ni = 0; ni < 2; ++ni) {
        acc[mi][ni] = __builtin_amdgcn_wmma_f32_16x16x32_f16(
            false, ah[mi], false, bh[ni], (short)0, acc[mi][ni], false, false);
        acc[mi][ni] = __builtin_amdgcn_wmma_f32_16x16x32_f16(
            false, ah[mi], false, bl[ni], (short)0, acc[mi][ni], false, false);
        acc[mi][ni] = __builtin_amdgcn_wmma_f32_16x16x32_f16(
            false, al[mi], false, bh[ni], (short)0, acc[mi][ni], false, false);
      }
  }

  // C/D layout: VGPR v -> M = v + hihalf*8, N = l15
#pragma unroll
  for (int mi = 0; mi < 2; ++mi) {
    const int ib = rw + (mi << 4) + (hihalf << 3);
#pragma unroll
    for (int ni = 0; ni < 2; ++ni) {
      const int c  = cw + (ni << 4) + l15;
      const float cn = cnorm[c];
#pragma unroll
      for (int v = 0; v < 8; ++v) {
        const int i = ib + v;
        float d2 = cnorm[BUFN + i] + cn - 2.0f * acc[mi][ni][v];
        float dist = sqrtf(fmaxf(d2, EPSD));
        int j = c - i;
        if (j >= 0 && j < BUFN) windows[(size_t)i * BUFN + j] = dist;
      }
    }
  }
}

// ---------------------------------------------------------------------------
// Kernel 3: per-row mean of 50 smallest via 4-pass MSB radix select.
// All dist > 0, so uint bit order == value order. Row cached in LDS.
// ---------------------------------------------------------------------------
__global__ void k_topk(const float* __restrict__ windows, float* __restrict__ out) {
  const int row = blockIdx.x;
  __shared__ float    vals[BUFN];       // 16 KB
  __shared__ unsigned hist[256];
  __shared__ float    red[256];
  __shared__ unsigned s_prefix;
  __shared__ int      s_kth, s_below;

  const float* w = windows + (size_t)row * BUFN;
  for (int t = threadIdx.x; t < BUFN; t += 256) vals[t] = w[t];
  if (threadIdx.x == 0) { s_prefix = 0u; s_kth = TOPK; s_below = 0; }
  __syncthreads();

  for (int pass = 0; pass < 4; ++pass) {
    const int shift = 24 - 8 * pass;
    hist[threadIdx.x] = 0u;
    __syncthreads();
    const unsigned pref   = s_prefix;
    const unsigned maskHi = (pass == 0) ? 0u : (0xFFFFFFFFu << (shift + 8));
    for (int t = threadIdx.x; t < BUFN; t += 256) {
      unsigned b = __float_as_uint(vals[t]);
      if ((b & maskHi) == (pref & maskHi))
        atomicAdd(&hist[(b >> shift) & 0xFF], 1u);
    }
    __syncthreads();
    if (threadIdx.x == 0) {
      unsigned cum = 0; int kth = s_kth; int bin = 0;
      for (; bin < 256; ++bin) {
        unsigned cb = hist[bin];
        if (cum + cb >= (unsigned)kth) break;
        cum += cb;
      }
      s_prefix |= ((unsigned)bin) << shift;
      s_kth     = kth - (int)cum;
      s_below  += (int)cum;
    }
    __syncthreads();
  }

  const unsigned kbits = s_prefix;            // exact bits of 50th smallest
  const float    kval  = __uint_as_float(kbits);
  const int      below = s_below;             // # strictly smaller
  float s = 0.f;
  for (int t = threadIdx.x; t < BUFN; t += 256) {
    unsigned b = __float_as_uint(vals[t]);
    if (b < kbits) s += vals[t];
  }
  red[threadIdx.x] = s;
  __syncthreads();
  for (int st = 128; st > 0; st >>= 1) {
    if (threadIdx.x < (unsigned)st) red[threadIdx.x] += red[threadIdx.x + st];
    __syncthreads();
  }
  if (threadIdx.x == 0) {
    float total = red[0] + (float)(TOPK - below) * kval;
    out[(size_t)row * 11 + 10] = total / (float)TOPK;
  }
}

// ---------------------------------------------------------------------------
// CNN kernels (tiny FLOPs -> plain VALU).
// ---------------------------------------------------------------------------
__global__ void k_conv1_pool(const float* __restrict__ x, const float* __restrict__ w1,
                             const float* __restrict__ b1, float* __restrict__ h1) {
  int idx = blockIdx.x * blockDim.x + threadIdx.x;
  if (idx >= BQ * 12 * 12 * 4) return;
  const int f  = idx & 3;
  int t = idx >> 2;
  const int ox = t % 12; t /= 12;
  const int oy = t % 12;
  const int n  = t / 12;
  float wr[25];
#pragma unroll
  for (int i = 0; i < 25; ++i) wr[i] = w1[i * 4 + f];
  const float bias = b1[f];
  const float* xi = x + (size_t)n * DIMD;
  float m = -INFINITY;
#pragma unroll
  for (int py = 0; py < 2; ++py)
#pragma unroll
    for (int px = 0; px < 2; ++px) {
      const int y0 = 2 * oy + py, x0 = 2 * ox + px;
      float s = bias;
#pragma unroll
      for (int ky = 0; ky < 5; ++ky)
#pragma unroll
        for (int kx = 0; kx < 5; ++kx)
          s += xi[(y0 + ky) * 28 + (x0 + kx)] * wr[ky * 5 + kx];
      m = fmaxf(m, s);
    }
  h1[idx] = fmaxf(m, 0.f);
}

__global__ void k_conv2_pool(const float* __restrict__ h1, const float* __restrict__ w2,
                             const float* __restrict__ b2, float* __restrict__ h2) {
  int idx = blockIdx.x * blockDim.x + threadIdx.x;
  if (idx >= BQ * 4 * 4 * 10) return;
  const int f  = idx % 10;
  int t = idx / 10;
  const int ox = t & 3; t >>= 2;
  const int oy = t & 3;
  const int n  = t >> 2;
  const float bias = b2[f];
  const float* hn = h1 + (size_t)n * 12 * 12 * 4;
  float m = -INFINITY;
  for (int py = 0; py < 2; ++py)
    for (int px = 0; px < 2; ++px) {
      const int y0 = 2 * oy + py, x0 = 2 * ox + px;
      float s = bias;
      for (int ky = 0; ky < 5; ++ky)
        for (int kx = 0; kx < 5; ++kx) {
          const float* hp = hn + ((y0 + ky) * 12 + (x0 + kx)) * 4;
          const float* wp = w2 + ((ky * 5 + kx) * 4) * 10 + f;
#pragma unroll
          for (int ci = 0; ci < 4; ++ci)
            s += hp[ci] * wp[ci * 10];
        }
      m = fmaxf(m, s);
    }
  h2[idx] = fmaxf(m, 0.f);
}

__global__ void k_dense1(const float* __restrict__ h2, const float* __restrict__ wd1,
                         const float* __restrict__ bd1, float* __restrict__ h3) {
  int idx = blockIdx.x * blockDim.x + threadIdx.x;
  if (idx >= BQ * 100) return;
  const int j = idx % 100;
  const int n = idx / 100;
  const float* h = h2 + (size_t)n * 160;
  float s = bd1[j];
  for (int i = 0; i < 160; ++i) s += h[i] * wd1[i * 100 + j];
  h3[idx] = fmaxf(s, 0.f);
}

__global__ void k_dense2_softmax(const float* __restrict__ h3, const float* __restrict__ wl,
                                 const float* __restrict__ bl, float* __restrict__ out) {
  int n = blockIdx.x * blockDim.x + threadIdx.x;
  if (n >= BQ) return;
  const float* h = h3 + (size_t)n * 100;
  float lg[10];
#pragma unroll
  for (int j = 0; j < 10; ++j) lg[j] = bl[j];
  for (int i = 0; i < 100; ++i) {
    const float hv = h[i];
    const float* wr = wl + i * 10;
#pragma unroll
    for (int j = 0; j < 10; ++j) lg[j] += hv * wr[j];
  }
  float m = lg[0];
#pragma unroll
  for (int j = 1; j < 10; ++j) m = fmaxf(m, lg[j]);
  float e[10], sum = 0.f;
#pragma unroll
  for (int j = 0; j < 10; ++j) { e[j] = expf(lg[j] - m); sum += e[j]; }
  const float inv = 1.0f / sum;
  float* o = out + (size_t)n * 11;
#pragma unroll
  for (int j = 0; j < 10; ++j) o[j] = e[j] * inv;
}

// ---------------------------------------------------------------------------
extern "C" void kernel_launch(void* const* d_in, const int* in_sizes, int n_in,
                              void* d_out, int out_size, void* d_ws, size_t ws_size,
                              hipStream_t stream) {
  (void)in_sizes; (void)n_in; (void)out_size; (void)ws_size;
  const float* x   = (const float*)d_in[0];
  const float* buf = (const float*)d_in[1];
  const float* w1  = (const float*)d_in[2];
  const float* b1  = (const float*)d_in[3];
  const float* w2  = (const float*)d_in[4];
  const float* b2  = (const float*)d_in[5];
  const float* wd1 = (const float*)d_in[6];
  const float* bd1 = (const float*)d_in[7];
  const float* wl  = (const float*)d_in[8];
  const float* bl  = (const float*)d_in[9];
  // d_in[10] = k (always 50, compiled in)
  float* out = (float*)d_out;

  // workspace carve-up (all offsets 256B aligned)
  char* ws = (char*)d_ws;
  const size_t sz_chi  = (size_t)CATP * KP * sizeof(_Float16);  // 13,516,800
  const size_t sz_norm = ((size_t)CATP * sizeof(float) + 255) & ~(size_t)255;
  _Float16* chi     = (_Float16*)(ws);
  _Float16* clo     = (_Float16*)(ws + sz_chi);
  float*    cnorm   = (float*)(ws + 2 * sz_chi);
  float*    windows = (float*)(ws + 2 * sz_chi + sz_norm);
  // CNN intermediates alias the windows region (read fully before conv1 runs)
  float* h1 = windows;                                          // 9,437,184 B
  float* h2 = h1 + (size_t)BQ * 12 * 12 * 4;                    // 2,621,440 B
  float* h3 = h2 + (size_t)BQ * 4 * 4 * 10;                     // 1,638,400 B

  // 1) convert + norms (includes zero padding rows up to CATP)
  k_convert<<<dim3(CATP), dim3(256), 0, stream>>>(x, buf, chi, clo, cnorm);
  // 2) banded WMMA distance GEMM -> windows (grid: 33 col-blocks x 64 row-blocks)
  k_band_gemm<<<dim3(33, BQ / 64), dim3(256), 0, stream>>>(chi, clo, cnorm, windows);
  // 3) top-50 mean per row -> out[:,10]
  k_topk<<<dim3(BQ), dim3(256), 0, stream>>>(windows, out);
  // 4) CNN (h1/h2/h3 reuse windows space; stream order makes this safe)
  k_conv1_pool<<<dim3((BQ * 12 * 12 * 4 + 255) / 256), dim3(256), 0, stream>>>(x, w1, b1, h1);
  k_conv2_pool<<<dim3((BQ * 4 * 4 * 10 + 255) / 256), dim3(256), 0, stream>>>(h1, w2, b2, h2);
  k_dense1<<<dim3((BQ * 100 + 255) / 256), dim3(256), 0, stream>>>(h2, wd1, bd1, h3);
  k_dense2_softmax<<<dim3((BQ + 63) / 64), dim3(64), 0, stream>>>(h3, wl, bl, out);
}
